// Attention_49838800503121
// MI455X (gfx1250) — matrix operand order repaired and checked by reading, not compiled, not run on hardware
//
#include <hip/hip_runtime.h>
#include <hip/hip_bf16.h>

typedef __bf16 bf16;
typedef __attribute__((ext_vector_type(16))) __bf16 v16bf;
typedef __attribute__((ext_vector_type(8)))  float  v8f;

#define B_   4
#define S_   4096
#define D_   1024
#define H_   16
#define HD_  64
#define M_   256
#define BH_  (B_ * H_)

// ---------------------------------------------------------------------------
// WMMA fragment helpers (CDNA5 16x16x32 bf16, wave32, contiguous k order, k = 16*(lane/16) + j)
// ---------------------------------------------------------------------------

// A matrix 16x32 (MxK), row-major source with leading dim lda.
// lane<16: row=lane,    K = k0+[0..15]
// lane>=16: row=lane-16, K = k0+[16..31]
static __device__ inline v16bf load_a16(const bf16* A, int lda, int row0, int k0, int lane) {
  int r  = row0 + (lane & 15);
  int kb = k0 + ((lane >> 4) << 4);
  const bf16* p = A + (long)r * lda + kb;
  v16bf a;
#pragma unroll
  for (int j = 0; j < 16; ++j) a[j] = p[j];
  return a;
}

// B matrix 32x16 (KxN), loaded from N-major ("B-transposed") storage BT[N][K],
// leading dim ldb. lane<16: col=lane, K=k0+[0..15]; lane>=16: K=k0+[16..31].
static __device__ inline v16bf load_bT16(const bf16* BT, int ldb, int col0, int k0, int lane) {
  int n  = col0 + (lane & 15);
  int kb = k0 + ((lane >> 4) << 4);
  const bf16* p = BT + (long)n * ldb + kb;
  v16bf b;
#pragma unroll
  for (int j = 0; j < 16; ++j) b[j] = p[j];
  return b;
}

static __device__ inline v8f wmma_bf16(v16bf a, v16bf b, v8f c) {
  // (neg_a, A, neg_b, B, c_mod, C, reuse_a, reuse_b)
  return __builtin_amdgcn_wmma_f32_16x16x32_bf16(false, a, false, b, (short)0, c,
                                                 false, false);
}

// CDNA5 async global->LDS copy (ASYNCcnt-tracked, no VGPR round-trip).
// Low 32 bits of a generic LDS pointer == wave-relative LDS byte offset.
static __device__ inline void async_stage16B(bf16* lds, const bf16* gsrc) {
  unsigned loff = (unsigned)(unsigned long long)(void*)lds;
  unsigned long long gaddr = (unsigned long long)gsrc;
  asm volatile("global_load_async_to_lds_b128 %0, %1, off"
               :: "v"(loff), "v"(gaddr) : "memory");
}

static __device__ inline void wait_async0() {
  asm volatile("s_wait_asynccnt 0x0" ::: "memory");
}

static __device__ inline void atomicMaxFloat(float* a, float v) {
  if (v >= 0.0f) atomicMax((int*)a, __float_as_int(v));
  else           atomicMin((unsigned int*)a, __float_as_uint(v));
}

// ---------------------------------------------------------------------------
// Conversion / init kernels
// ---------------------------------------------------------------------------

__global__ void cvt_bf16_kernel(const float* __restrict__ src, bf16* __restrict__ dst, long n) {
  long i = (long)blockIdx.x * blockDim.x + threadIdx.x;
  if (i < n) dst[i] = (bf16)src[i];
}

// W is (K=1024 x N=1024) row-major f32; emit Wt[N][K] bf16.
__global__ void transpose_bf16_kernel(const float* __restrict__ W, bf16* __restrict__ Wt) {
  int i = blockIdx.x * blockDim.x + threadIdx.x; // i = n*1024 + k
  int n = i >> 10, k = i & 1023;
  Wt[i] = (bf16)W[(long)k * D_ + n];
}

__global__ void init_kmax_kernel(float* kmax) {
  if (threadIdx.x < BH_) kmax[threadIdx.x] = -1e30f;
}

// ---------------------------------------------------------------------------
// QKV projection GEMM: C[row, col] = sum_k X[row,k] * W[k,col] + bias[col]
// Block = 256 threads (8 waves); block tile 128 rows x 64 cols; K-step 32.
// B tiles double-buffered in LDS via async global->LDS copies.
// mode 0: Q -> Qx[bh][s][hd] * HD^-0.5
// mode 1: K -> Kx[bh][s][hd] * HD^-0.5 * mask
// mode 2: V -> Vt[bh][hd][s] * mask        (head-transposed for kv GEMM)
// ---------------------------------------------------------------------------
__global__ __launch_bounds__(256) void qkv_gemm(const bf16* __restrict__ Xb,
                                                const bf16* __restrict__ Wt,
                                                const float* __restrict__ bias,
                                                const float* __restrict__ mask,
                                                bf16* __restrict__ out, int mode) {
  __shared__ bf16 sB[2][64 * 32]; // double-buffered 64 cols x 32 k (2 x 4 KB)
  int tid = threadIdx.x, lane = tid & 31, w = tid >> 5;
  int row0 = blockIdx.x * 128 + w * 16;
  int col0 = blockIdx.y * 64;

  // per-thread staging slot: one 16B async B128 each
  int e  = tid * 8;
  int cc = e >> 5, kk = e & 31;
  const bf16* gB = Wt + (long)(col0 + cc) * D_ + kk; // advance by k0

  async_stage16B(&sB[0][e], gB);
  wait_async0();
  __syncthreads();

  v8f c0 = {0,0,0,0,0,0,0,0}, c1 = c0, c2 = c0, c3 = c0;
  int cur = 0;
  for (int k0 = 0; k0 < D_; k0 += 32) {
    if (k0 + 32 < D_) async_stage16B(&sB[cur ^ 1][e], gB + k0 + 32);
    if (k0 + 64 < D_)
      __builtin_prefetch(Xb + (long)(row0 + (lane & 15)) * D_ + k0 + 64, 0, 0);

    v16bf a = load_a16(Xb, D_, row0, k0, lane);
    c0 = wmma_bf16(a, load_bT16(sB[cur], 32,  0, 0, lane), c0);
    c1 = wmma_bf16(a, load_bT16(sB[cur], 32, 16, 0, lane), c1);
    c2 = wmma_bf16(a, load_bT16(sB[cur], 32, 32, 0, lane), c2);
    c3 = wmma_bf16(a, load_bT16(sB[cur], 32, 48, 0, lane), c3);

    wait_async0();      // next buffer landed (per-wave ASYNCcnt)
    __syncthreads();    // all waves done reading cur + done staging
    cur ^= 1;
  }

  v8f acc[4] = {c0, c1, c2, c3};
  int rbase = row0 + ((lane >> 4) << 3);
#pragma unroll
  for (int nt = 0; nt < 4; ++nt) {
    int colabs = col0 + nt * 16 + (lane & 15);
    float bv = bias[colabs];
    int h = colabs >> 6, hd = colabs & 63;
#pragma unroll
    for (int r = 0; r < 8; ++r) {
      int row = rbase + r;              // row == b*S + s
      int b = row >> 12, s = row & (S_ - 1);
      int bh = b * H_ + h;
      float v = acc[nt][r] + bv;
      if (mode == 0) {
        out[((long)bh * S_ + s) * HD_ + hd] = (bf16)(v * 0.125f);
      } else if (mode == 1) {
        out[((long)bh * S_ + s) * HD_ + hd] = (bf16)(v * 0.125f * mask[row]);
      } else {
        out[((long)(bh * HD_ + hd)) * S_ + s] = (bf16)(v * mask[row]);
      }
    }
  }
}

// ---------------------------------------------------------------------------
// phi tile: u[s, m] = sum_k Arow[s,k] * proj[m,k] for a 32-row tile,
// written to LDS (32 x 256 f32). Block = 64 threads (2 waves).
// ---------------------------------------------------------------------------
static __device__ inline void phi_u_tile(const bf16* __restrict__ Arows,
                                         const bf16* __restrict__ projb,
                                         int s0, float* uLDS) {
  int tid = threadIdx.x, lane = tid & 31, w = tid >> 5;
  int row0 = s0 + w * 16;
  v8f z = {0,0,0,0,0,0,0,0};
  v8f c[16];
#pragma unroll
  for (int i = 0; i < 16; ++i) c[i] = z;

  for (int k0 = 0; k0 < HD_; k0 += 32) {
    v16bf a = load_a16(Arows, HD_, row0, k0, lane);
#pragma unroll
    for (int nt = 0; nt < 16; ++nt) {
      v16bf b = load_bT16(projb, HD_, nt * 16, k0, lane);
      c[nt] = wmma_bf16(a, b, c[nt]);
    }
  }
  int lr0 = w * 16 + ((lane >> 4) << 3);
#pragma unroll
  for (int nt = 0; nt < 16; ++nt) {
    int col = nt * 16 + (lane & 15);
#pragma unroll
    for (int r = 0; r < 8; ++r) uLDS[(lr0 + r) * M_ + col] = c[nt][r];
  }
}

// Pass 1 for K: global per-(b,h) max of u via LDS reduce + sign-split atomic max.
__global__ __launch_bounds__(64) void kstat_kernel(const bf16* __restrict__ Kx,
                                                   const bf16* __restrict__ projb,
                                                   float* __restrict__ kmax) {
  __shared__ float uLDS[32 * M_]; // 32 KB
  int bh = blockIdx.y, s0 = blockIdx.x * 32;
  phi_u_tile(Kx + (long)bh * S_ * HD_, projb, s0, uLDS);
  __syncthreads();
  float m = -1e30f;
  for (int i = threadIdx.x; i < 32 * M_; i += 64) m = fmaxf(m, uLDS[i]);
  __syncthreads();
  uLDS[threadIdx.x] = m;
  __syncthreads();
  for (int off = 32; off > 0; off >>= 1) {
    if (threadIdx.x < off)
      uLDS[threadIdx.x] = fmaxf(uLDS[threadIdx.x], uLDS[threadIdx.x + off]);
    __syncthreads();
  }
  if (threadIdx.x == 0) atomicMaxFloat(&kmax[bh], uLDS[0]);
}

// Q feature map: per-row max stabilizer, diag from Qx row, write q' bf16 (S x M).
__global__ __launch_bounds__(64) void qexp_kernel(const bf16* __restrict__ Qx,
                                                  const bf16* __restrict__ projb,
                                                  bf16* __restrict__ qp) {
  __shared__ float uLDS[32 * M_];
  int bh = blockIdx.y, s0 = blockIdx.x * 32;
  const bf16* A = Qx + (long)bh * S_ * HD_;
  phi_u_tile(A, projb, s0, uLDS);
  __syncthreads();
  int t = threadIdx.x;
  if (t < 32) {
    int s = s0 + t;
    float d = 0.f;
    const bf16* xr = A + (long)s * HD_;
    for (int k = 0; k < HD_; ++k) { float x = (float)xr[k]; d += x * x; }
    d *= 0.5f;
    float st = -1e30f;
    for (int m = 0; m < M_; ++m) st = fmaxf(st, uLDS[t * M_ + m]);
    bf16* orow = qp + ((long)bh * S_ + s) * M_;
    for (int m = 0; m < M_; ++m)
      orow[m] = (bf16)(0.0625f * (expf(uLDS[t * M_ + m] - d - st) + 1e-4f));
  }
}

// K feature map: global stabilizer, write k' TRANSPOSED (M x S) for kv GEMM.
__global__ __launch_bounds__(64) void kexp_kernel(const bf16* __restrict__ Kx,
                                                  const bf16* __restrict__ projb,
                                                  const float* __restrict__ kmax,
                                                  bf16* __restrict__ kpT) {
  __shared__ float uLDS[32 * M_];
  int bh = blockIdx.y, s0 = blockIdx.x * 32;
  const bf16* A = Kx + (long)bh * S_ * HD_;
  phi_u_tile(A, projb, s0, uLDS);
  __syncthreads();
  int t = threadIdx.x;
  if (t < 32) {
    int s = s0 + t;
    float d = 0.f;
    const bf16* xr = A + (long)s * HD_;
    for (int k = 0; k < HD_; ++k) { float x = (float)xr[k]; d += x * x; }
    d *= 0.5f;
    float st = kmax[bh];
    for (int m = 0; m < M_; ++m)
      kpT[((long)bh * M_ + m) * S_ + s] =
          (bf16)(0.0625f * (expf(uLDS[t * M_ + m] - d - st) + 1e-4f));
  }
}

// ksum[bh][m] = sum_s k'[s][m]  (row-sum of kpT)
__global__ __launch_bounds__(256) void ksum_kernel(const bf16* __restrict__ kpT,
                                                   float* __restrict__ ksum) {
  __shared__ float red[256];
  int bm = blockIdx.x; // bh*M + m
  const bf16* row = kpT + (long)bm * S_;
  float s = 0.f;
  for (int i = threadIdx.x; i < S_; i += 256) s += (float)row[i];
  red[threadIdx.x] = s;
  __syncthreads();
  for (int off = 128; off > 0; off >>= 1) {
    if (threadIdx.x < off) red[threadIdx.x] += red[threadIdx.x + off];
    __syncthreads();
  }
  if (threadIdx.x == 0) ksum[bm] = red[0];
}

// kv[m, d] = sum_s k'T[m,s] * Vt[d,s]; output kvT[d][m] bf16 (for out GEMM).
// B tiles double-buffered in LDS via async global->LDS copies.
__global__ __launch_bounds__(256) void kv_gemm(const bf16* __restrict__ kpT,
                                               const bf16* __restrict__ Vt,
                                               bf16* __restrict__ kvT) {
  __shared__ bf16 sB[2][64 * 32];
  int bh = blockIdx.y;
  int tid = threadIdx.x, lane = tid & 31, w = tid >> 5;
  int m0 = blockIdx.x * 128 + w * 16;
  const bf16* A  = kpT + (long)bh * M_ * S_;
  const bf16* BT = Vt  + (long)bh * HD_ * S_;

  int e  = tid * 8;
  int cc = e >> 5, kk = e & 31;
  const bf16* gB = BT + (long)cc * S_ + kk; // advance by k0

  async_stage16B(&sB[0][e], gB);
  wait_async0();
  __syncthreads();

  v8f c0 = {0,0,0,0,0,0,0,0}, c1 = c0, c2 = c0, c3 = c0;
  int cur = 0;
  for (int k0 = 0; k0 < S_; k0 += 32) {
    if (k0 + 32 < S_) async_stage16B(&sB[cur ^ 1][e], gB + k0 + 32);
    if (k0 + 64 < S_)
      __builtin_prefetch(A + (long)(m0 + (lane & 15)) * S_ + k0 + 64, 0, 0);

    v16bf a = load_a16(A, S_, m0, k0, lane);
    c0 = wmma_bf16(a, load_bT16(sB[cur], 32,  0, 0, lane), c0);
    c1 = wmma_bf16(a, load_bT16(sB[cur], 32, 16, 0, lane), c1);
    c2 = wmma_bf16(a, load_bT16(sB[cur], 32, 32, 0, lane), c2);
    c3 = wmma_bf16(a, load_bT16(sB[cur], 32, 48, 0, lane), c3);

    wait_async0();
    __syncthreads();
    cur ^= 1;
  }
  v8f acc[4] = {c0, c1, c2, c3};
  int rbase = m0 + ((lane >> 4) << 3);
#pragma unroll
  for (int nt = 0; nt < 4; ++nt) {
    int d = nt * 16 + (lane & 15);
#pragma unroll
    for (int r = 0; r < 8; ++r) {
      int m = rbase + r;
      kvT[((long)bh * HD_ + d) * M_ + m] = (bf16)acc[nt][r];
    }
  }
}

// denom[bh*S+s] = sum_m q'[s,m] * ksum[bh,m]
__global__ __launch_bounds__(256) void denom_kernel(const bf16* __restrict__ qp,
                                                    const float* __restrict__ ksum,
                                                    float* __restrict__ denom) {
  long idx = (long)blockIdx.x * blockDim.x + threadIdx.x; // bh*S + s
  int bh = (int)(idx >> 12);
  const bf16* row = qp + idx * M_;
  const float* ks = ksum + (long)bh * M_;
  float acc = 0.f;
  for (int m = 0; m < M_; ++m) acc += (float)row[m] * ks[m];
  denom[idx] = acc;
}

// out[b, s, h*64+d] = z[s] * sum_m q'[s,m] * kv[m,d];
// whole 32 KB B panel async-staged into LDS once per block.
__global__ __launch_bounds__(256) void out_gemm(const bf16* __restrict__ qp,
                                                const bf16* __restrict__ kvT,
                                                const float* __restrict__ denom,
                                                float* __restrict__ out) {
  __shared__ bf16 sB[HD_ * M_]; // 32 KB: kvT[d][m] for this head
  int bh = blockIdx.y;
  int tid = threadIdx.x, lane = tid & 31, w = tid >> 5;
  int s0 = blockIdx.x * 128 + w * 16;
  const bf16* A  = qp  + (long)bh * S_ * M_;
  const bf16* BT = kvT + (long)bh * HD_ * M_;

  { // 64 elements (128 B) per thread = 8 async B128s
    int e = tid * 64;
#pragma unroll
    for (int j = 0; j < 8; ++j) async_stage16B(&sB[e + j * 8], BT + e + j * 8);
    wait_async0();
  }
  __syncthreads();

  v8f c0 = {0,0,0,0,0,0,0,0}, c1 = c0, c2 = c0, c3 = c0;
  for (int k0 = 0; k0 < M_; k0 += 32) {
    v16bf a = load_a16(A, M_, s0, k0, lane);
    c0 = wmma_bf16(a, load_bT16(sB, M_,  0, k0, lane), c0);
    c1 = wmma_bf16(a, load_bT16(sB, M_, 16, k0, lane), c1);
    c2 = wmma_bf16(a, load_bT16(sB, M_, 32, k0, lane), c2);
    c3 = wmma_bf16(a, load_bT16(sB, M_, 48, k0, lane), c3);
  }

  int b = bh >> 4, h = bh & 15;
  int rbase = s0 + ((lane >> 4) << 3);
  float z[8];
#pragma unroll
  for (int r = 0; r < 8; ++r)
    z[r] = 1.0f / (denom[(long)bh * S_ + rbase + r] + 1e-6f);

  v8f acc[4] = {c0, c1, c2, c3};
#pragma unroll
  for (int nt = 0; nt < 4; ++nt) {
    int col = nt * 16 + (lane & 15);
#pragma unroll
    for (int r = 0; r < 8; ++r) {
      int s = rbase + r;
      out[((long)(b * S_ + s)) * D_ + h * HD_ + col] = acc[nt][r] * z[r];
    }
  }
}

// ---------------------------------------------------------------------------
// Host-side launcher
// ---------------------------------------------------------------------------
extern "C" void kernel_launch(void* const* d_in, const int* in_sizes, int n_in,
                              void* d_out, int out_size, void* d_ws, size_t ws_size,
                              hipStream_t stream) {
  const float* X    = (const float*)d_in[0];
  const float* mask = (const float*)d_in[1];
  const float* Wq   = (const float*)d_in[2];
  const float* bq   = (const float*)d_in[3];
  const float* Wk   = (const float*)d_in[4];
  const float* bk   = (const float*)d_in[5];
  const float* Wv   = (const float*)d_in[6];
  const float* bv   = (const float*)d_in[7];
  const float* proj = (const float*)d_in[8];
  float* out = (float*)d_out;
  (void)in_sizes; (void)n_in; (void)out_size; (void)ws_size;

  char* ws = (char*)d_ws;
  size_t off = 0;
  auto alloc = [&](size_t bytes) -> void* {
    off = (off + 255) & ~(size_t)255;
    void* p = ws + off;
    off += bytes;
    return p;
  };

  bf16*  Xb    = (bf16*)alloc((size_t)B_ * S_ * D_ * 2);
  bf16*  Wtq   = (bf16*)alloc((size_t)D_ * D_ * 2);
  bf16*  Wtk   = (bf16*)alloc((size_t)D_ * D_ * 2);
  bf16*  Wtv   = (bf16*)alloc((size_t)D_ * D_ * 2);
  bf16*  projb = (bf16*)alloc((size_t)M_ * HD_ * 2);
  bf16*  Qx    = (bf16*)alloc((size_t)BH_ * S_ * HD_ * 2);
  bf16*  Kx    = (bf16*)alloc((size_t)BH_ * S_ * HD_ * 2);
  bf16*  Vt    = (bf16*)alloc((size_t)BH_ * HD_ * S_ * 2);
  bf16*  qp    = (bf16*)alloc((size_t)BH_ * S_ * M_ * 2);
  bf16*  kpT   = (bf16*)alloc((size_t)BH_ * M_ * S_ * 2);
  float* kmax  = (float*)alloc((size_t)BH_ * 4);
  float* ksum  = (float*)alloc((size_t)BH_ * M_ * 4);
  bf16*  kvT   = (bf16*)alloc((size_t)BH_ * HD_ * M_ * 2);
  float* denom = (float*)alloc((size_t)BH_ * S_ * 4);

  // precision conversions / layout transforms
  cvt_bf16_kernel<<<(B_ * S_ * D_) / 256, 256, 0, stream>>>(X, Xb, (long)B_ * S_ * D_);
  transpose_bf16_kernel<<<(D_ * D_) / 256, 256, 0, stream>>>(Wq, Wtq);
  transpose_bf16_kernel<<<(D_ * D_) / 256, 256, 0, stream>>>(Wk, Wtk);
  transpose_bf16_kernel<<<(D_ * D_) / 256, 256, 0, stream>>>(Wv, Wtv);
  cvt_bf16_kernel<<<(M_ * HD_) / 256, 256, 0, stream>>>(proj, projb, (long)M_ * HD_);

  // QKV projection GEMMs (WMMA + async LDS staging)
  dim3 g1(128, 16);
  qkv_gemm<<<g1, 256, 0, stream>>>(Xb, Wtq, bq, mask, Qx, 0);
  qkv_gemm<<<g1, 256, 0, stream>>>(Xb, Wtk, bk, mask, Kx, 1);
  qkv_gemm<<<g1, 256, 0, stream>>>(Xb, Wtv, bv, mask, Vt, 2);

  // feature maps
  init_kmax_kernel<<<1, 64, 0, stream>>>(kmax);
  dim3 gphi(S_ / 32, BH_);
  kstat_kernel<<<gphi, 64, 0, stream>>>(Kx, projb, kmax);
  qexp_kernel<<<gphi, 64, 0, stream>>>(Qx, projb, qp);
  kexp_kernel<<<gphi, 64, 0, stream>>>(Kx, projb, kmax, kpT);
  ksum_kernel<<<BH_ * M_, 256, 0, stream>>>(kpT, ksum);

  // kv and output GEMMs (WMMA)
  kv_gemm<<<dim3(2, BH_), 256, 0, stream>>>(kpT, Vt, kvT);
  denom_kernel<<<(BH_ * S_) / 256, 256, 0, stream>>>(qp, ksum, denom);
  out_gemm<<<dim3(S_ / 128, BH_), 256, 0, stream>>>(qp, kvT, denom, out);
}
